// MixedAttention_35923106464002
// MI455X (gfx1250) — matrix operand order repaired and checked by reading, not compiled, not run on hardware
//
#include <hip/hip_runtime.h>

typedef __attribute__((ext_vector_type(16))) __bf16 v16bf;
typedef __attribute__((ext_vector_type(8)))  __bf16 v8bf;
typedef __attribute__((ext_vector_type(4)))  __bf16 v4bf;
typedef __attribute__((ext_vector_type(8)))  float  v8f;

#define TOK   2048
#define CDIM  512
#define BATCH 4
#define NHEAD 8
#define DHEAD 64

__device__ __forceinline__ v8f wmma_bf16(v16bf a, v16bf b, v8f c) {
  // D = A(16x32 bf16) * B(32x16 bf16) + C(16x16 f32)
  return __builtin_amdgcn_wmma_f32_16x16x32_bf16(false, a, false, b, (short)0, c,
                                                 false, false);
}

// ---- CDNA5 async global->LDS copy (ASYNCcnt-tracked), 16 bytes per lane ----
__device__ __forceinline__ void async_load_b128(unsigned lds_off, const void* gp) {
  unsigned long long ga = (unsigned long long)gp;
  asm volatile("global_load_async_to_lds_b128 %0, %1, off"
               :: "v"(lds_off), "v"(ga) : "memory");
}
__device__ __forceinline__ void wait_async0() {
  asm volatile("s_wait_asynccnt 0x0" ::: "memory");
}
__device__ __forceinline__ void wait_ds0() {
  asm volatile("s_wait_dscnt 0x0" ::: "memory");
}

// ---- VALU-pipe xor reduction within 16-lane halves via DPP16 ROW_XMASK ----
template <int CTRL>
__device__ __forceinline__ float dpp_max_step(float v) {
  int t = __builtin_amdgcn_update_dpp(__builtin_bit_cast(int, v),
                                      __builtin_bit_cast(int, v),
                                      CTRL, 0xf, 0xf, true);
  return fmaxf(v, __builtin_bit_cast(float, t));
}
__device__ __forceinline__ float hmax16_dpp(float v) {
  v = dpp_max_step<0x161>(v);  // row_xmask: lane ^ 1
  v = dpp_max_step<0x162>(v);  // lane ^ 2
  v = dpp_max_step<0x164>(v);  // lane ^ 4
  v = dpp_max_step<0x168>(v);  // lane ^ 8
  return v;
}

__device__ __forceinline__ v16bf ld_frag16(const __bf16* p) {
  v16bf f;
#pragma unroll
  for (int e = 0; e < 16; ++e) f[e] = p[e];
  return f;
}

// ---------------------------------------------------------------------------
// Tiled GEMM: Out[M,N] = A[M,K] @ W[K,N] (+ bias), bf16 WMMA, f32 acc.
// Block = 256 thr (8 waves). Tile: BM=128 (16 rows/wave), BN=64, BK=64.
// bf16 A tiles staged via async global->LDS DMA; 8 WMMAs per barrier.
// ---------------------------------------------------------------------------
template <bool A_BF16, bool BIAS_F32OUT>
__global__ __launch_bounds__(256) void gemm_tile(const void* __restrict__ Ap,
                                                 const float* __restrict__ W,
                                                 const float* __restrict__ bias,
                                                 void* __restrict__ Out,
                                                 int M, int N, int K) {
  __shared__ __align__(16) __bf16 a_s[128][64];  // [m][k]
  __shared__ __align__(16) __bf16 b_s[64][64];   // [n][k] (W transposed tile)

  const int tid  = threadIdx.x;
  const int wave = tid >> 5;
  const int lane = tid & 31;
  const int l15  = lane & 15;
  const int hi   = (lane >= 16);
  const int m0   = blockIdx.y * 128;
  const int n0   = blockIdx.x * 64;

  v8f c[4];
#pragma unroll
  for (int t = 0; t < 4; ++t)
#pragma unroll
    for (int r = 0; r < 8; ++r) c[t][r] = 0.0f;

  for (int k0 = 0; k0 < K; k0 += 64) {
    if (A_BF16) {
      // Async DMA: 1024 x 16B chunks, 4 per thread (ASYNCcnt-tracked).
#pragma unroll
      for (int cc = 0; cc < 4; ++cc) {
        int ch = tid + cc * 256;
        int row = ch >> 3, off = (ch & 7) * 8;
        async_load_b128((unsigned)(size_t)&a_s[row][off],
                        &((const __bf16*)Ap)[(size_t)(m0 + row) * K + k0 + off]);
      }
    } else {
#pragma unroll
      for (int it = 0; it < 8; ++it) {
        int i = tid * 4 + it * 1024;
        int row = i >> 6, kk = i & 63;
        float4 v = *(const float4*)&((const float*)Ap)[(size_t)(m0 + row) * K + k0 + kk];
        v4bf o;
        o[0] = (__bf16)v.x; o[1] = (__bf16)v.y; o[2] = (__bf16)v.z; o[3] = (__bf16)v.w;
        *(v4bf*)&a_s[row][kk] = o;
      }
    }
#pragma unroll
    for (int it = 0; it < 4; ++it) {
      int i = tid * 4 + it * 1024;
      int kk = i >> 6, n = i & 63;
      float4 w = *(const float4*)&W[(size_t)(k0 + kk) * N + n0 + n];
      b_s[n + 0][kk] = (__bf16)w.x;
      b_s[n + 1][kk] = (__bf16)w.y;
      b_s[n + 2][kk] = (__bf16)w.z;
      b_s[n + 3][kk] = (__bf16)w.w;
    }
    if (A_BF16) wait_async0();
    __syncthreads();

    // Two A fragments (k 0..31 / 32..63).
    const __bf16* ar = &a_s[wave * 16 + l15][0];
    const int kb = hi ? 16 : 0;
    v16bf a0, a1;
#pragma unroll
    for (int e = 0; e < 8; ++e) {
      a0[e]     = ar[kb + e];
      a0[e + 8] = ar[kb + 8 + e];
      a1[e]     = ar[32 + kb + e];
      a1[e + 8] = ar[32 + kb + 8 + e];
    }

#pragma unroll
    for (int t = 0; t < 4; ++t) {
      // Preload both B fragments, then issue both WMMAs.
      v16bf bf0 = ld_frag16(&b_s[t * 16 + l15][kb]);
      v16bf bf1 = ld_frag16(&b_s[t * 16 + l15][32 + kb]);
      c[t] = wmma_bf16(a0, bf0, c[t]);
      c[t] = wmma_bf16(a1, bf1, c[t]);
    }
    __syncthreads();
  }

#pragma unroll
  for (int t = 0; t < 4; ++t)
#pragma unroll
    for (int r = 0; r < 8; ++r) {
      int row = m0 + wave * 16 + (hi ? r + 8 : r);
      int col = n0 + t * 16 + l15;
      if (BIAS_F32OUT)
        ((float*)Out)[(size_t)row * N + col] = c[t][r] + bias[col];
      else
        ((__bf16*)Out)[(size_t)row * N + col] = (__bf16)c[t][r];
    }
}

// ---------------------------------------------------------------------------
// Per-(tensor, batch, channel) max & sum over token dim.
// ---------------------------------------------------------------------------
__global__ __launch_bounds__(128) void pool_kernel(const __bf16* __restrict__ Q,
                                                   const __bf16* __restrict__ K,
                                                   const __bf16* __restrict__ V,
                                                   float* __restrict__ pmax,
                                                   float* __restrict__ psum) {
  const int t = blockIdx.z;
  const int b = blockIdx.y;
  const int c = blockIdx.x * 128 + threadIdx.x;
  const __bf16* src = (t == 0) ? Q : (t == 1) ? K : V;
  src += (size_t)b * TOK * CDIM;
  float mx = -3.0e38f, sm = 0.0f;
  for (int n = 0; n < TOK; ++n) {
    float v = (float)src[(size_t)n * CDIM + c];
    mx = fmaxf(mx, v);
    sm += v;
  }
  pmax[(t * BATCH + b) * CDIM + c] = mx;
  psum[(t * BATCH + b) * CDIM + c] = sm;
}

// ---------------------------------------------------------------------------
// Channel-attention MLP: scale = sigmoid(relu((max+avg) @ w1) @ w2).
// ---------------------------------------------------------------------------
__global__ __launch_bounds__(512) void ca_kernel(const float* __restrict__ w1,
                                                 const float* __restrict__ w2,
                                                 const float* __restrict__ pmax,
                                                 const float* __restrict__ psum,
                                                 float* __restrict__ scales) {
  __shared__ float s[CDIM];
  __shared__ float hid[32];
  const int tb = blockIdx.x;
  const int tid = threadIdx.x;
  s[tid] = pmax[tb * CDIM + tid] + psum[tb * CDIM + tid] * (1.0f / (float)TOK);
  __syncthreads();
  if (tid < 32) {
    float a = 0.0f;
    for (int cc = 0; cc < CDIM; ++cc) a += s[cc] * w1[cc * 32 + tid];
    hid[tid] = fmaxf(a, 0.0f);
  }
  __syncthreads();
  float a = 0.0f;
#pragma unroll
  for (int j = 0; j < 32; ++j) a += hid[j] * w2[j * CDIM + tid];
  scales[tb * CDIM + tid] = 1.0f / (1.0f + __expf(-a));
}

// ---------------------------------------------------------------------------
// Flash attention: grid = (TOK/128, NHEAD, BATCH), block = 256 (8 waves).
// BJ=64: 18 WMMAs per j-iter, softmax bookkeeping amortized over 64 columns.
// sc_k folded into Q; sc_v applied at epilogue; log2(e) folded -> native exp2.
// K tile staged via async global->LDS DMA (raw bf16).
// ---------------------------------------------------------------------------
__global__ __launch_bounds__(256) void attn_kernel(const __bf16* __restrict__ Qb,
                                                   const __bf16* __restrict__ Kb,
                                                   const __bf16* __restrict__ Vb,
                                                   const float* __restrict__ scales,
                                                   __bf16* __restrict__ Ob) {
  __shared__ __align__(16) __bf16 lds_q[8][16][DHEAD];  // per-wave Q tile
  __shared__ __align__(16) __bf16 lds_k[64][DHEAD];     // K tile [j][d] (raw)
  __shared__ __align__(16) __bf16 lds_vt[DHEAD][64];    // V tile transposed [d][j]
  __shared__ __align__(16) __bf16 lds_p[8][16][64];     // per-wave probs [i][j]

  const int tid  = threadIdx.x;
  const int wave = tid >> 5;
  const int lane = tid & 31;
  const int l15  = lane & 15;
  const int hi   = (lane >= 16);
  const int b    = blockIdx.z;
  const int h    = blockIdx.y;
  const int i0   = blockIdx.x * 128 + wave * 16;

  const float* sc_q = scales + (0 * BATCH + b) * CDIM;
  const float* sc_k = scales + (1 * BATCH + b) * CDIM;
  const float* sc_v = scales + (2 * BATCH + b) * CDIM;
  // 512^-0.5 * log2(e): scores land directly in log2 domain for native exp2.
  const float qscale = 0.044194173824159216f * 1.4426950408889634f;

  // Stage this wave's Q rows with all multiplicative factors folded in.
  for (int i = lane * 8; i < 16 * DHEAD; i += 32 * 8) {
    int r = i >> 6, d = i & 63;
    int c = h * DHEAD + d;
    v8bf qv = *(const v8bf*)&Qb[((size_t)(b * TOK + i0 + r)) * CDIM + c];
    v8bf o;
#pragma unroll
    for (int e = 0; e < 8; ++e)
      o[e] = (__bf16)((float)qv[e] * sc_q[c + e] * sc_k[c + e] * qscale);
    *(v8bf*)&lds_q[wave][r][d] = o;
  }
  __syncthreads();

  // Two A fragments covering d = 0..31 and 32..63.
  const int kh = hi ? 8 : 0;
  const int kb = hi ? 16 : 0;
  v16bf aq0, aq1;
  {
    const __bf16* qr = &lds_q[wave][l15][0];
#pragma unroll
    for (int e = 0; e < 8; ++e) {
      aq0[e]     = qr[kb + e];
      aq0[e + 8] = qr[kb + 8 + e];
      aq1[e]     = qr[32 + kb + e];
      aq1[e + 8] = qr[32 + kb + 8 + e];
    }
  }

  v16bf onesf;
#pragma unroll
  for (int e = 0; e < 16; ++e) onesf[e] = (__bf16)1.0f;

  float m_r[8], l_r[8];
  v8f acc[4];
#pragma unroll
  for (int r = 0; r < 8; ++r) { m_r[r] = -3.0e38f; l_r[r] = 0.0f; }
#pragma unroll
  for (int t = 0; t < 4; ++t)
#pragma unroll
    for (int r = 0; r < 8; ++r) acc[t][r] = 0.0f;

  for (int j0 = 0; j0 < TOK; j0 += 64) {
    // Stage K (async DMA) and V^T (vector load + transpose scatter): 64x64 tiles.
#pragma unroll
    for (int cc = 0; cc < 2; ++cc) {
      int ch = tid + cc * 256;
      int r8 = ch >> 3, d8 = (ch & 7) * 8;
      async_load_b128((unsigned)(size_t)&lds_k[r8][d8],
                      &Kb[((size_t)(b * TOK + j0 + r8)) * CDIM + h * DHEAD + d8]);
    }
#pragma unroll
    for (int cc = 0; cc < 2; ++cc) {
      int ch = tid + cc * 256;
      int r8 = ch >> 3, d8 = (ch & 7) * 8;
      v8bf vv = *(const v8bf*)&Vb[((size_t)(b * TOK + j0 + r8)) * CDIM + h * DHEAD + d8];
#pragma unroll
      for (int e = 0; e < 8; ++e) lds_vt[d8 + e][r8] = vv[e];
    }
    wait_async0();
    __syncthreads();

    // Scores: s[16 i][64 j] = Q tile @ K^T tile (log2 domain).
    v8f s[4];
#pragma unroll
    for (int jj = 0; jj < 4; ++jj) {
      v16bf bk0 = ld_frag16(&lds_k[jj * 16 + l15][kb]);
      v16bf bk1 = ld_frag16(&lds_k[jj * 16 + l15][32 + kb]);
      v8f z;
#pragma unroll
      for (int r = 0; r < 8; ++r) z[r] = 0.0f;
      z = wmma_bf16(aq0, bk0, z);
      s[jj] = wmma_bf16(aq1, bk1, z);
    }

    // Online softmax: row max on VALU pipe (DPP row_xmask), native exp2.
    float alpha[8];
#pragma unroll
    for (int r = 0; r < 8; ++r) {
      float smx = fmaxf(fmaxf(s[0][r], s[1][r]), fmaxf(s[2][r], s[3][r]));
      smx = hmax16_dpp(smx);
      float mn = fmaxf(m_r[r], smx);
      alpha[r] = exp2f(m_r[r] - mn);
      m_r[r]   = mn;
      int row  = hi ? (r + 8) : r;
#pragma unroll
      for (int jj = 0; jj < 4; ++jj)
        lds_p[wave][row][jj * 16 + l15] = (__bf16)exp2f(s[jj][r] - mn);
    }
#pragma unroll
    for (int t = 0; t < 4; ++t)
#pragma unroll
      for (int r = 0; r < 8; ++r) acc[t][r] *= alpha[r];

    wait_ds0();

    // Probs as two A fragments (j 0..31 / 32..63).
    v16bf pa0, pa1;
    {
      const __bf16* pr = &lds_p[wave][l15][0];
#pragma unroll
      for (int e = 0; e < 8; ++e) {
        pa0[e]     = pr[kb + e];
        pa0[e + 8] = pr[kb + 8 + e];
        pa1[e]     = pr[32 + kb + e];
        pa1[e + 8] = pr[32 + kb + 8 + e];
      }
    }
    // Row sums via WMMA against all-ones B (no cross-lane shuffles).
    {
      v8f rs;
#pragma unroll
      for (int r = 0; r < 8; ++r) rs[r] = 0.0f;
      rs = wmma_bf16(pa0, onesf, rs);
      rs = wmma_bf16(pa1, onesf, rs);
#pragma unroll
      for (int r = 0; r < 8; ++r) l_r[r] = l_r[r] * alpha[r] + rs[r];
    }
    // acc += P @ V
#pragma unroll
    for (int t = 0; t < 4; ++t) {
      v16bf bv0 = ld_frag16(&lds_vt[t * 16 + l15][kb]);
      v16bf bv1 = ld_frag16(&lds_vt[t * 16 + l15][32 + kb]);
      acc[t] = wmma_bf16(pa0, bv0, acc[t]);
      acc[t] = wmma_bf16(pa1, bv1, acc[t]);
    }
    __syncthreads();
  }

  // Epilogue: normalize, apply channel-attention V-scale per output column.
#pragma unroll
  for (int t = 0; t < 4; ++t)
#pragma unroll
    for (int r = 0; r < 8; ++r) {
      int row = i0 + (hi ? r + 8 : r);
      int col = h * DHEAD + t * 16 + l15;
      Ob[((size_t)(b * TOK + row)) * CDIM + col] =
          (__bf16)(acc[t][r] / l_r[r] * sc_v[col]);
    }
}

// ---------------------------------------------------------------------------
extern "C" void kernel_launch(void* const* d_in, const int* in_sizes, int n_in,
                              void* d_out, int out_size, void* d_ws, size_t ws_size,
                              hipStream_t stream) {
  (void)in_sizes; (void)n_in; (void)out_size; (void)ws_size;

  const float* x  = (const float*)d_in[0];
  const float* m  = (const float*)d_in[1];
  const float* Wq = (const float*)d_in[2];
  const float* Wk = (const float*)d_in[3];
  const float* Wv = (const float*)d_in[4];
  const float* Wo = (const float*)d_in[5];
  const float* bo = (const float*)d_in[6];
  const float* w1 = (const float*)d_in[7];
  const float* w2 = (const float*)d_in[8];
  float* out = (float*)d_out;

  char* ws = (char*)d_ws;
  const size_t TBYTES = (size_t)BATCH * TOK * CDIM * sizeof(__bf16);  // 8 MiB
  __bf16* Qb = (__bf16*)(ws + 0 * TBYTES);
  __bf16* Kb = (__bf16*)(ws + 1 * TBYTES);
  __bf16* Vb = (__bf16*)(ws + 2 * TBYTES);
  __bf16* Ab = (__bf16*)(ws + 3 * TBYTES);
  float* pmax   = (float*)(ws + 4 * TBYTES);
  float* psum   = pmax + 3 * BATCH * CDIM;
  float* scales = psum + 3 * BATCH * CDIM;

  const int M = BATCH * TOK;  // 8192
  dim3 g_gemm(CDIM / 64, M / 128);

  gemm_tile<false, false><<<g_gemm, 256, 0, stream>>>(x, Wq, nullptr, Qb, M, CDIM, CDIM);
  gemm_tile<false, false><<<g_gemm, 256, 0, stream>>>(m, Wk, nullptr, Kb, M, CDIM, CDIM);
  gemm_tile<false, false><<<g_gemm, 256, 0, stream>>>(m, Wv, nullptr, Vb, M, CDIM, CDIM);

  pool_kernel<<<dim3(CDIM / 128, BATCH, 3), 128, 0, stream>>>(Qb, Kb, Vb, pmax, psum);
  ca_kernel<<<12, 512, 0, stream>>>(w1, w2, pmax, psum, scales);

  attn_kernel<<<dim3(TOK / 128, NHEAD, BATCH), 256, 0, stream>>>(Qb, Kb, Vb, scales, Ab);

  gemm_tile<true, true><<<g_gemm, 256, 0, stream>>>(Ab, Wo, bo, out, M, CDIM, CDIM);
}
